// EdgeUpdaterModule_27041114095823
// MI455X (gfx1250) — compile-verified, matrix operand order checked by reading, not run on hardware
//
#include <hip/hip_runtime.h>
#include <hip/hip_bf16.h>

#define NODE_DIM 128
#define EDGE_DIM 64
#define HIDDEN   128
#define IN_DIM   320
#define LN_EPS   1e-5f

typedef __bf16 bf16x2  __attribute__((ext_vector_type(2)));
typedef __bf16 bf16x4  __attribute__((ext_vector_type(4)));
typedef __bf16 bf16x8  __attribute__((ext_vector_type(8)));
typedef __bf16 bf16x16 __attribute__((ext_vector_type(16)));
typedef float  f32x8   __attribute__((ext_vector_type(8)));
typedef int    i32x4   __attribute__((ext_vector_type(4)));

// LDS layout (bf16 elements unless noted)
#define W1T_OFF 0                        // [128 cols][320 k]
#define WGT_OFF (128 * IN_DIM)           // 40960
#define W2T_OFF (WGT_OFF + 128 * IN_DIM) // 81920   [128][128]
#define WRT_OFF (W2T_OFF + 128 * HIDDEN) // 98304   [128][64]
#define WTOTAL  (WRT_OFF + 128 * EDGE_DIM) // 106496 bf16 = 212992 B
#define CAT_ELEMS (8 * 16 * IN_DIM)      // 8 waves x [16][320] = 40960
#define PAR_OFF_B ((WTOTAL + CAT_ELEMS) * 2) // 294912 B, fp32 params after
#define SMEM_BYTES (PAR_OFF_B + 8 * 128 * 4) // 299008 B

#if defined(__HIP_DEVICE_COMPILE__) && \
    __has_builtin(__builtin_amdgcn_global_load_async_to_lds_b128)
#define HAVE_ASYNC_LDS 1
#else
#define HAVE_ASYNC_LDS 0
#endif

#if defined(__HIP_DEVICE_COMPILE__) && \
    __has_builtin(__builtin_amdgcn_s_wait_asynccnt)
#define ASYNC_WAIT() __builtin_amdgcn_s_wait_asynccnt(0)
#else
#define ASYNC_WAIT() asm volatile("s_wait_asynccnt 0x0" ::: "memory")
#endif

// float -> bf16 round-to-nearest-even
static __device__ __forceinline__ __bf16 f2bf(float f) {
  unsigned u = __builtin_bit_cast(unsigned, f);
  unsigned short r = (unsigned short)((u + 0x7FFFu + ((u >> 16) & 1u)) >> 16);
  return __builtin_bit_cast(__bf16, r);
}

// tanh-form GELU on the hardware transcendental pipe (~8 VALU vs ~35 for erff)
static __device__ __forceinline__ float gelu_fast(float x) {
  float p = x * x;
  float u = 0.7978845608f * x * __builtin_fmaf(0.044715f, p, 1.0f);
  float e = __expf(2.0f * u);
  return x - x * __builtin_amdgcn_rcpf(e + 1.0f);  // x*(1 - 1/(1+e^{2u}))
}
static __device__ __forceinline__ float sigmoid_fast(float y) {
  return __builtin_amdgcn_rcpf(1.0f + __expf(-y));
}

// One 16x16x32 bf16 A/B fragment from an LDS row pointer (ISA 7.12.2 layout):
// lane<16 holds K = kb..kb+7 , kb+16..kb+23 ; lane>=16 shifts kb by +8 (hsel).
static __device__ __forceinline__ bf16x16 ld_frag(const __bf16* rowp, int kb) {
  bf16x8 lo = *(const bf16x8*)(rowp + kb);
  bf16x8 hi = *(const bf16x8*)(rowp + kb + 16);
  bf16x16 r;
#pragma unroll
  for (int i = 0; i < 8; ++i) { r[i] = lo[i]; r[i + 8] = hi[i]; }
  return r;
}

static __device__ __forceinline__ f32x8 wmma_bf16(bf16x16 a, bf16x16 b, f32x8 c) {
  return __builtin_amdgcn_wmma_f32_16x16x32_bf16(false, a, false, b, (short)0, c,
                                                 false, false);
}

static __device__ __forceinline__ void zero8(f32x8* a) {
#pragma unroll
  for (int n = 0; n < 8; ++n)
#pragma unroll
    for (int v = 0; v < 8; ++v) a[n][v] = 0.0f;
}

// ---- prep kernel: fp32 weights -> bf16, transposed to [col][k], in d_ws ----
extern "C" __global__ void weight_prep_kernel(const float* __restrict__ W1,
                                              const float* __restrict__ Wg,
                                              const float* __restrict__ W2,
                                              const float* __restrict__ Wr,
                                              __bf16* __restrict__ ws) {
  int i0 = blockIdx.x * blockDim.x + threadIdx.x;
  int st = gridDim.x * blockDim.x;
  for (int idx = i0; idx < IN_DIM * 128; idx += st) {
    int k = idx >> 7, c = idx & 127;
    ws[W1T_OFF + c * IN_DIM + k] = f2bf(W1[idx]);
    ws[WGT_OFF + c * IN_DIM + k] = f2bf(Wg[idx]);
  }
  for (int idx = i0; idx < HIDDEN * 128; idx += st) {
    int k = idx >> 7, c = idx & 127;
    ws[W2T_OFF + c * HIDDEN + k] = f2bf(W2[idx]);
  }
  for (int idx = i0; idx < EDGE_DIM * 128; idx += st) {
    int k = idx >> 7, c = idx & 127;
    ws[WRT_OFF + c * EDGE_DIM + k] = f2bf(Wr[idx]);
  }
}

extern "C" __global__ void __launch_bounds__(256, 1)
edge_updater_kernel(const float* __restrict__ x, const int* __restrict__ ei,
                    const float* __restrict__ ea,
                    const float* __restrict__ W1, const float* __restrict__ b1,
                    const float* __restrict__ g1, const float* __restrict__ be1,
                    const float* __restrict__ W2, const float* __restrict__ b2,
                    const float* __restrict__ Wg, const float* __restrict__ bg,
                    const float* __restrict__ Wr, const float* __restrict__ br,
                    const float* __restrict__ g2, const float* __restrict__ be2,
                    const __bf16* __restrict__ wsT, int preconv,
                    float* __restrict__ out, int E, int tiles) {
  extern __shared__ __align__(16) char smem[];
  __bf16* lds = (__bf16*)smem;
  __bf16* W1T = lds + W1T_OFF;
  __bf16* WgT = lds + WGT_OFF;
  __bf16* W2T = lds + W2T_OFF;
  __bf16* WrT = lds + WRT_OFF;
  float*  par = (float*)(smem + PAR_OFF_B);  // [8][128]: b1 g1 be1 bg b2 br g2 be2

  const int tid  = threadIdx.x;
  const int lane = tid & 31;
  const int wave = tid >> 5;
  const int l15  = lane & 15;
  const int hsel = (lane >> 4) << 3;  // K-select (A/B frag) == row-base (C/D frag)

  // ---- stage weights into LDS ----
  if (preconv) {
#if HAVE_ASYNC_LDS
    // gfx1250 async copy: global (bf16, pre-transposed) -> LDS, ASYNCcnt-tracked
    char* src = (char*)wsT;
    char* dst = (char*)smem;
    for (int off = tid * 16; off < WTOTAL * 2; off += 256 * 16) {
      __builtin_amdgcn_global_load_async_to_lds_b128(
          (__attribute__((address_space(1))) i32x4*)(src + off),
          (__attribute__((address_space(3))) i32x4*)(dst + off), 0, 0);
    }
    ASYNC_WAIT();
#else
    for (int i = tid * 8; i < WTOTAL; i += 256 * 8)
      *(bf16x8*)(lds + i) = *(const bf16x8*)(wsT + i);
#endif
  } else {  // fallback: convert fp32 weights in-kernel
    for (int idx = tid; idx < IN_DIM * 128; idx += blockDim.x) {
      int k = idx >> 7, c = idx & 127;
      W1T[c * IN_DIM + k] = f2bf(W1[idx]);
      WgT[c * IN_DIM + k] = f2bf(Wg[idx]);
    }
    for (int idx = tid; idx < HIDDEN * 128; idx += blockDim.x) {
      int k = idx >> 7, c = idx & 127;
      W2T[c * HIDDEN + k] = f2bf(W2[idx]);
    }
    for (int idx = tid; idx < EDGE_DIM * 128; idx += blockDim.x) {
      int k = idx >> 7, c = idx & 127;
      WrT[c * EDGE_DIM + k] = f2bf(Wr[idx]);
    }
  }
  for (int i = tid; i < 128; i += blockDim.x) {
    par[0 * 128 + i] = b1[i];  par[1 * 128 + i] = g1[i];
    par[2 * 128 + i] = be1[i]; par[3 * 128 + i] = bg[i];
    par[4 * 128 + i] = b2[i];  par[5 * 128 + i] = br[i];
    par[6 * 128 + i] = g2[i];  par[7 * 128 + i] = be2[i];
  }
  __syncthreads();

  // per-wave staging: cat = [16][320]; h aliases cat cols 0..127 (dead after GEMM1)
  __bf16* cat = lds + WTOTAL + wave * 16 * IN_DIM;
  __bf16* hA  = cat;  // [16][IN_DIM] rows, first 128 cols reused for h

  for (int t = blockIdx.x; t < tiles; t += gridDim.x) {
    const int e0 = t * 128 + wave * 16;  // this wave's 16-edge M-tile

    // ---- gather cat = [x[src] | x[dst] | edge_attr] into LDS as bf16 ----
#pragma unroll 1
    for (int r = 0; r < 16; ++r) {
      int e = e0 + r;
      __bf16* crow = cat + r * IN_DIM;
      if (e < E) {
        int s = ei[e], d = ei[E + e];
        float4 xs = ((const float4*)(x + (long)s * NODE_DIM))[lane];
        float4 xd = ((const float4*)(x + (long)d * NODE_DIM))[lane];
        float2 ev = ((const float2*)(ea + (long)e * EDGE_DIM))[lane];
        bf16x4 a; a[0]=f2bf(xs.x); a[1]=f2bf(xs.y); a[2]=f2bf(xs.z); a[3]=f2bf(xs.w);
        bf16x4 b; b[0]=f2bf(xd.x); b[1]=f2bf(xd.y); b[2]=f2bf(xd.z); b[3]=f2bf(xd.w);
        bf16x2 c; c[0]=f2bf(ev.x); c[1]=f2bf(ev.y);
        *(bf16x4*)(crow + lane * 4)               = a;
        *(bf16x4*)(crow + NODE_DIM + lane * 4)    = b;
        *(bf16x2*)(crow + 2 * NODE_DIM + lane * 2) = c;
      } else {
        __bf16 z = f2bf(0.0f);
        bf16x4 za; za[0]=z; za[1]=z; za[2]=z; za[3]=z;
        bf16x2 zb; zb[0]=z; zb[1]=z;
        *(bf16x4*)(crow + lane * 4)               = za;
        *(bf16x4*)(crow + NODE_DIM + lane * 4)    = za;
        *(bf16x2*)(crow + 2 * NODE_DIM + lane * 2) = zb;
      }
    }
    asm volatile("s_wait_dscnt 0x0" ::: "memory");  // per-wave LDS W->R ordering

    // ---- GEMM1: acc1 = cat@W1, accG = cat@Wg (shared A fragments) ----
    f32x8 acc1[8], accG[8];
    zero8(acc1); zero8(accG);
#pragma unroll 1
    for (int kk = 0; kk < IN_DIM / 32; ++kk) {
      bf16x16 a = ld_frag(cat + l15 * IN_DIM, kk * 32 + hsel);
#pragma unroll
      for (int n = 0; n < 8; ++n) {
        int col = n * 16 + l15;
        bf16x16 bw1 = ld_frag(W1T + col * IN_DIM, kk * 32 + hsel);
        acc1[n] = wmma_bf16(a, bw1, acc1[n]);
        bf16x16 bwg = ld_frag(WgT + col * IN_DIM, kk * 32 + hsel);
        accG[n] = wmma_bf16(a, bwg, accG[n]);
      }
    }

    // ---- GEMM3: res = edge_attr @ Wr (cat cols 256..319) ----
    f32x8 accR[8];
    zero8(accR);
#pragma unroll 1
    for (int kk = 0; kk < EDGE_DIM / 32; ++kk) {
      bf16x16 a = ld_frag(cat + l15 * IN_DIM + 2 * NODE_DIM, kk * 32 + hsel);
#pragma unroll
      for (int n = 0; n < 8; ++n) {
        bf16x16 bw = ld_frag(WrT + (n * 16 + l15) * EDGE_DIM, kk * 32 + hsel);
        accR[n] = wmma_bf16(a, bw, accR[n]);
      }
    }

    // ---- LayerNorm1 (+b1) -> GELU -> h (bf16, aliased onto cat cols 0..127) ----
    float s[8], q[8], mean[8], rstd[8];
#pragma unroll
    for (int v = 0; v < 8; ++v) { s[v] = 0.f; q[v] = 0.f; }
#pragma unroll
    for (int n = 0; n < 8; ++n) {
      float bb = par[0 * 128 + n * 16 + l15];
#pragma unroll
      for (int v = 0; v < 8; ++v) {
        float tv = acc1[n][v] + bb;
        acc1[n][v] = tv;
        s[v] += tv; q[v] += tv * tv;
      }
    }
#pragma unroll
    for (int v = 0; v < 8; ++v)
#pragma unroll
      for (int m = 8; m >= 1; m >>= 1) {  // reduce within each 16-lane half
        s[v] += __shfl_xor(s[v], m, 32);
        q[v] += __shfl_xor(q[v], m, 32);
      }
#pragma unroll
    for (int v = 0; v < 8; ++v) {
      mean[v] = s[v] * (1.0f / HIDDEN);
      float var = q[v] * (1.0f / HIDDEN) - mean[v] * mean[v];
      rstd[v] = rsqrtf(var + LN_EPS);
    }
#pragma unroll
    for (int n = 0; n < 8; ++n) {
      int col = n * 16 + l15;
      float gg = par[1 * 128 + col], bb = par[2 * 128 + col];
#pragma unroll
      for (int v = 0; v < 8; ++v) {
        float tv = (acc1[n][v] - mean[v]) * rstd[v] * gg + bb;
        hA[(v + hsel) * IN_DIM + col] = f2bf(gelu_fast(tv));
      }
    }
    // ---- gate = sigmoid(accG + bg) in place ----
#pragma unroll
    for (int n = 0; n < 8; ++n) {
      float bb = par[3 * 128 + n * 16 + l15];
#pragma unroll
      for (int v = 0; v < 8; ++v)
        accG[n][v] = sigmoid_fast(accG[n][v] + bb);
    }
    asm volatile("s_wait_dscnt 0x0" ::: "memory");

    // ---- GEMM2: update = h @ W2 (reuse acc1) ----
    zero8(acc1);
#pragma unroll 1
    for (int kk = 0; kk < HIDDEN / 32; ++kk) {
      bf16x16 a = ld_frag(hA + l15 * IN_DIM, kk * 32 + hsel);
#pragma unroll
      for (int n = 0; n < 8; ++n) {
        bf16x16 bw = ld_frag(W2T + (n * 16 + l15) * HIDDEN, kk * 32 + hsel);
        acc1[n] = wmma_bf16(a, bw, acc1[n]);
      }
    }

    // ---- combine: f = (update+b2)*gate + (res+br), LayerNorm2, store ----
#pragma unroll
    for (int v = 0; v < 8; ++v) { s[v] = 0.f; q[v] = 0.f; }
#pragma unroll
    for (int n = 0; n < 8; ++n) {
      int col = n * 16 + l15;
      float b2v = par[4 * 128 + col], brv = par[5 * 128 + col];
#pragma unroll
      for (int v = 0; v < 8; ++v) {
        float f = (acc1[n][v] + b2v) * accG[n][v] + (accR[n][v] + brv);
        acc1[n][v] = f;
        s[v] += f; q[v] += f * f;
      }
    }
#pragma unroll
    for (int v = 0; v < 8; ++v)
#pragma unroll
      for (int m = 8; m >= 1; m >>= 1) {
        s[v] += __shfl_xor(s[v], m, 32);
        q[v] += __shfl_xor(q[v], m, 32);
      }
#pragma unroll
    for (int v = 0; v < 8; ++v) {
      mean[v] = s[v] * (1.0f / HIDDEN);
      float var = q[v] * (1.0f / HIDDEN) - mean[v] * mean[v];
      rstd[v] = rsqrtf(var + LN_EPS);
    }
#pragma unroll
    for (int n = 0; n < 8; ++n) {
      int col = n * 16 + l15;
      float gg = par[6 * 128 + col], bb = par[7 * 128 + col];
#pragma unroll
      for (int v = 0; v < 8; ++v) {
        int e = e0 + v + hsel;
        if (e < E)
          out[(long)e * HIDDEN + col] =
              (acc1[n][v] - mean[v]) * rstd[v] * gg + bb;
      }
    }
  }
}

extern "C" void kernel_launch(void* const* d_in, const int* in_sizes, int n_in,
                              void* d_out, int out_size, void* d_ws, size_t ws_size,
                              hipStream_t stream) {
  const float* x   = (const float*)d_in[0];
  const int*   ei  = (const int*)d_in[1];
  const float* ea  = (const float*)d_in[2];
  const float* W1  = (const float*)d_in[3];
  const float* b1  = (const float*)d_in[4];
  const float* g1  = (const float*)d_in[5];
  const float* be1 = (const float*)d_in[6];
  const float* W2  = (const float*)d_in[7];
  const float* b2  = (const float*)d_in[8];
  const float* Wg  = (const float*)d_in[9];
  const float* bg  = (const float*)d_in[10];
  const float* Wr  = (const float*)d_in[11];
  const float* br  = (const float*)d_in[12];
  const float* g2  = (const float*)d_in[13];
  const float* be2 = (const float*)d_in[14];

  const int E = in_sizes[2] / EDGE_DIM;     // 500000 edges
  const int tiles = (E + 127) / 128;        // 128 edges per block-iteration
  int grid = tiles < 2048 ? tiles : 2048;   // persistent blocks

  const int preconv = (ws_size >= (size_t)WTOTAL * 2) ? 1 : 0;
  if (preconv)
    weight_prep_kernel<<<128, 256, 0, stream>>>(W1, Wg, W2, Wr, (__bf16*)d_ws);

  edge_updater_kernel<<<grid, 256, SMEM_BYTES, stream>>>(
      x, ei, ea, W1, b1, g1, be1, W2, b2, Wg, bg, Wr, br, g2, be2,
      (const __bf16*)d_ws, preconv, (float*)d_out, E, tiles);
}
